// Embedder_71846212928262
// MI455X (gfx1250) — matrix operand order repaired and checked by reading, not compiled, not run on hardware
//
#include <hip/hip_runtime.h>
#include <hip/hip_bf16.h>

// ---------------- problem constants ----------------
#define BATCH   32768
#define EMB     2048
#define KDIM    320          // 5*8*8 pooled features
#define KSTEPS  10           // 320 / 32
#define MTILES  (BATCH/16)   // 2048
#define NTILES  (EMB/16)     // 128
#define KACT    103          // ceil(0.05*2048)

typedef __attribute__((ext_vector_type(16))) _Float16 v16h;
typedef __attribute__((ext_vector_type(8)))  float    v8f;

// ---- WMMA VGPR-layout packing (CDNA5 ISA 7.12.2) ----
// A-matrix 16x32 f16: lanes 0-15 hold K in {0..7}U{16..23}, lanes 16-31 hold
// K in {8..15}U{24..31}; halves 0..7 -> low-K group, 8..15 -> +16 group.
__device__ __forceinline__ int a_pack(int mrow, int k) {
    int lane = mrow + (((k & 15) >= 8) ? 16 : 0);
    int half = ((k >= 16) ? 8 : 0) + (k & 7);
    return lane * 16 + half;           // index within 512-half tile
}
// B-matrix 32x16 f16: same K order as A: lanes 0-15 hold K in {0..7}U{16..23}, lanes 16-31 K in {8..15}U{24..31}.
__device__ __forceinline__ int b_pack(int k, int ncol) {
    int lane = ncol + (((k & 15) >= 8) ? 16 : 0);
    int half = ((k >= 16) ? 8 : 0) + (k & 7);
    return lane * 16 + half;
}

__device__ __forceinline__ unsigned mono(float f) {   // order-preserving f32->u32
    unsigned u = __float_as_uint(f);
    return (u & 0x80000000u) ? ~u : (u | 0x80000000u);
}

// ---------------- kernel 1: conv3x3 + BN + ReLU + maxpool3x3/3, pack A ----------------
__global__ void __launch_bounds__(320)
conv_pool_pack(const float* __restrict__ x, const float* __restrict__ conv_w,
               const float* __restrict__ bn_gamma, const float* __restrict__ bn_beta,
               const float* __restrict__ bn_mean, const float* __restrict__ bn_var,
               _Float16* __restrict__ Ap) {
    __shared__ float xs[28 * 28];
    const int b = blockIdx.x;
    const int t = threadIdx.x;
    for (int i = t; i < 784; i += 320) xs[i] = x[(size_t)b * 784 + i];
    __syncthreads();

    const int c  = t / 64;
    const int r  = t % 64;
    const int py = r / 8;
    const int px = r % 8;

    float w[9];
#pragma unroll
    for (int i = 0; i < 9; ++i) w[i] = conv_w[c * 9 + i];
    const float inv_std = bn_gamma[c] * rsqrtf(bn_var[c] + 1e-5f);
    const float mean = bn_mean[c], beta = bn_beta[c];

    float best = -INFINITY;
#pragma unroll
    for (int i = 0; i < 3; ++i) {
#pragma unroll
        for (int j = 0; j < 3; ++j) {
            const int y = py * 3 + i, xc = px * 3 + j;
            float acc = 0.f;
#pragma unroll
            for (int ky = 0; ky < 3; ++ky)
#pragma unroll
                for (int kx = 0; kx < 3; ++kx)
                    acc = fmaf(xs[(y + ky) * 28 + (xc + kx)], w[ky * 3 + kx], acc);
            float v = (acc - mean) * inv_std + beta;
            v = fmaxf(v, 0.f);
            best = fmaxf(best, v);
        }
    }

    const int K = c * 64 + py * 8 + px;
    const int mtile = b >> 4, mrow = b & 15;
    const int ks = K >> 5, kk = K & 31;
    Ap[(size_t)(mtile * KSTEPS + ks) * 512 + a_pack(mrow, kk)] = (_Float16)best;
}

// ---------------- kernel 2: fc_w f32 -> f16 packed B ----------------
__global__ void __launch_bounds__(256)
pack_weights(const float* __restrict__ fc_w, _Float16* __restrict__ Bp) {
    const int idx = blockIdx.x * 256 + threadIdx.x;
    const int n = idx / KDIM, k = idx % KDIM;
    const int ntile = n >> 4, ncol = n & 15;
    const int ks = k >> 5, kk = k & 31;
    Bp[(size_t)(ntile * KSTEPS + ks) * 512 + b_pack(kk, ncol)] = (_Float16)fc_w[idx];
}

// ---------------- kernel 3: fused WMMA GEMM + kWTA ----------------
// One block = one 16-row M-tile. z tile (16x2048 f32 = 128KB) lives entirely in
// LDS (CDNA5 320KB WGP LDS), kWTA radix-select runs on it, only binary output
// touches HBM. 8 waves; each wave covers 16 N-tiles (4 groups of 4), then owns
// 2 rows for the wave-synchronous 103rd-largest selection.
//
// Dynamic LDS layout:
//   zrow : 16*2048 f32   = 131072 B
//   Alds : 10*512  f16   =  10240 B
//   hist : 8*256   u32   =   8192 B   (per-wave histograms)
__global__ void __launch_bounds__(256)
fused_gemm_kwta(const _Float16* __restrict__ Ap, const _Float16* __restrict__ Bp,
                float* __restrict__ out) {
    extern __shared__ char smem_raw[];
    float*    zrow = (float*)smem_raw;                       // [16][2048]
    _Float16* Alds = (_Float16*)(smem_raw + 131072);         // [KSTEPS*512]
    unsigned* hist = (unsigned*)(smem_raw + 131072 + 10240); // [8*256]

    const int tid   = threadIdx.x;
    const int wave  = tid >> 5;
    const int lane  = tid & 31;
    const int mtile = blockIdx.x;

    // ---- stage packed A tile (16x320 f16 = 10KB) into LDS ----
    {
        const unsigned* Asrc = (const unsigned*)(Ap + (size_t)mtile * KSTEPS * 512);
        unsigned* Adst = (unsigned*)Alds;
#pragma unroll
        for (int i = 0; i < KSTEPS; ++i)          // 2560 dwords / 256 threads
            Adst[tid + i * 256] = Asrc[tid + i * 256];
    }
    __syncthreads();

    // ---- GEMM phase: wave covers N-tiles [wave*16, wave*16+16) ----
    const v16h* __restrict__ Bb = (const v16h*)Bp;
#pragma unroll
    for (int g = 0; g < 4; ++g) {
        const int nt0 = wave * 16 + g * 4;
        v8f acc[4];
#pragma unroll
        for (int i = 0; i < 4; ++i) { v8f zz = {}; acc[i] = zz; }
#pragma unroll
        for (int ks = 0; ks < KSTEPS; ++ks) {
            v16h a = *(const v16h*)&Alds[ks * 512 + lane * 16];   // ds_load
            if (ks + 1 < KSTEPS)   // global_prefetch_b8 for next B chunk
                __builtin_prefetch((const void*)&Bb[(size_t)(nt0 * KSTEPS + ks + 1) * 32 + lane], 0, 1);
#pragma unroll
            for (int t = 0; t < 4; ++t) {
                v16h b = Bb[(size_t)((nt0 + t) * KSTEPS + ks) * 32 + lane];
                acc[t] = __builtin_amdgcn_wmma_f32_16x16x32_f16(
                    false, a, false, b, (short)0, acc[t], false, false);
            }
        }
        // C/D layout: lanes 0-15 -> M=r, lanes 16-31 -> M=r+8; N = lane%16
        const int mrow0 = (lane < 16) ? 0 : 8;
        const int ncol  = lane & 15;
#pragma unroll
        for (int t = 0; t < 4; ++t)
#pragma unroll
            for (int r = 0; r < 8; ++r)
                zrow[(mrow0 + r) * EMB + (nt0 + t) * 16 + ncol] = acc[t][r];
    }
    __syncthreads();

    // ---- kWTA phase: wave owns rows 2*wave and 2*wave+1 ----
    const int wbase = wave * 256;
#pragma unroll 1
    for (int r2 = 0; r2 < 2; ++r2) {
        const int row = wave * 2 + r2;
        // cache the row's monotone keys in 64 VGPRs (2048 / 32 lanes)
        unsigned kc[64];
#pragma unroll
        for (int j = 0; j < 64; ++j)
            kc[j] = mono(zrow[row * EMB + lane + j * 32]);

        unsigned prefix = 0u;
        int kneed = KACT;
#pragma unroll 1
        for (int byte = 3; byte >= 0; --byte) {
            // zero this wave's 256 bins (8 per lane)
#pragma unroll
            for (int i = 0; i < 8; ++i) hist[wbase + lane * 8 + i] = 0u;
            __builtin_amdgcn_wave_barrier();
            // histogram of matching-prefix keys
#pragma unroll
            for (int j = 0; j < 64; ++j) {
                const unsigned key = kc[j];
                const bool match = (byte == 3) ||
                    ((key >> (8 * byte + 8)) == (prefix >> (8 * byte + 8)));
                if (match) atomicAdd(&hist[wbase + ((key >> (8 * byte)) & 0xFFu)], 1u);
            }
            __builtin_amdgcn_wave_barrier();
            // per-lane local bins + cross-lane inclusive suffix sum
            unsigned cnt[8], lsum = 0u;
#pragma unroll
            for (int i = 0; i < 8; ++i) { cnt[i] = hist[wbase + lane * 8 + i]; lsum += cnt[i]; }
            unsigned suf = lsum;
#pragma unroll
            for (int off = 1; off < 32; off <<= 1) {
                unsigned o = (unsigned)__shfl_down((int)suf, off, 32);
                if (lane + off < 32) suf += o;
            }
            unsigned run = suf - lsum;        // count in bins of strictly-higher lanes
            int sel = -1; unsigned newpref = 0u; int newk = 0;
#pragma unroll
            for (int i = 7; i >= 0; --i) {    // bin index = lane*8+i, descending
                const unsigned inc = run + cnt[i];
                if (sel < 0 && (int)inc >= kneed && (int)run < kneed) {
                    sel = lane * 8 + i;
                    newpref = prefix | ((unsigned)sel << (8 * byte));
                    newk = kneed - (int)run;
                }
                run = inc;
            }
            const unsigned long long bal = __ballot(sel >= 0);
            const int src = __ffsll(bal) - 1;
            prefix = (unsigned)__shfl((int)newpref, src, 32);
            kneed  = __shfl(newk, src, 32);
        }

        // binarize: z >= thr  <=>  key >= prefix (monotone map)
        const unsigned thr = prefix;
        float* outp = out + (size_t)(mtile * 16 + row) * EMB;
#pragma unroll
        for (int j = 0; j < 64; ++j)
            outp[lane + j * 32] = (kc[j] >= thr) ? 1.0f : 0.0f;
    }
}

// ---------------- launcher ----------------
extern "C" void kernel_launch(void* const* d_in, const int* in_sizes, int n_in,
                              void* d_out, int out_size, void* d_ws, size_t ws_size,
                              hipStream_t stream) {
    const float* x        = (const float*)d_in[0];
    const float* conv_w   = (const float*)d_in[1];
    const float* bn_gamma = (const float*)d_in[2];
    const float* bn_beta  = (const float*)d_in[3];
    const float* bn_mean  = (const float*)d_in[4];
    const float* bn_var   = (const float*)d_in[5];
    const float* fc_w     = (const float*)d_in[6];
    float* out = (float*)d_out;

    _Float16* Ap = (_Float16*)d_ws;                   // 32768*320 f16 = 20.97 MB
    _Float16* Bp = Ap + (size_t)BATCH * KDIM;         // 2048*320  f16 =  1.31 MB

    const size_t smem = 131072 + 10240 + 8192;        // 149,504 B < 320KB WGP LDS
    (void)hipFuncSetAttribute((const void*)fused_gemm_kwta,
                              hipFuncAttributeMaxDynamicSharedMemorySize, (int)smem);

    conv_pool_pack<<<BATCH, 320, 0, stream>>>(x, conv_w, bn_gamma, bn_beta, bn_mean, bn_var, Ap);
    pack_weights<<<(EMB * KDIM) / 256, 256, 0, stream>>>(fc_w, Bp);
    fused_gemm_kwta<<<MTILES, 256, smem, stream>>>(Ap, Bp, out);
}
